// FlowMiner_23063974379777
// MI455X (gfx1250) — compile-verified, matrix operand order checked by reading, not run on hardware
//
#include <hip/hip_runtime.h>
#include <hip/hip_bf16.h>

#define NN   50000
#define EE   800000
#define DIN  64
#define HID  128
#define HEADS 4
#define GG   64
#define NC   8
#define kEPS 1e-5f
#define SLOPE 0.2f
#define E2   (EE + NN)      // edges + self loops
#define MTILES (NN / 16)    // 3125 exact

typedef __attribute__((ext_vector_type(16))) _Float16 v16h;
typedef __attribute__((ext_vector_type(8)))  float    v8f;

// ---------------------------------------------------------------------------
// Weight pre-pack: f32 row-major W[k*ldw + c] -> f16 in WMMA B-fragment order.
// Fragment element (ks, tn, lane, j):
//   c  = tn*16 + (lane&15),  kb = (lane>>4)*8
//   k  = ks*32 + (j<8 ? kb+j : 16+kb+(j-8))
// Packed index = ((ks*8 + tn)*32 + lane)*16 + j   (contiguous v16h per lane)
// ---------------------------------------------------------------------------
__global__ void pack_w_f16(const float* __restrict__ W, int ldw, int K,
                           _Float16* __restrict__ Wp) {
    int idx = blockIdx.x * 256 + threadIdx.x;
    if (idx >= K * HID) return;
    int j    = idx & 15;
    int lane = (idx >> 4) & 31;
    int t    = idx >> 9;          // ks*8 + tn
    int tn   = t & 7, ks = t >> 3;
    int c  = tn * 16 + (lane & 15);
    int kb = (lane >> 4) << 3;
    int k  = (ks << 5) + ((j < 8) ? (kb + j) : (16 + kb + j - 8));
    Wp[idx] = (_Float16)W[(size_t)k * ldw + c];
}

// ---------------------------------------------------------------------------
// WMMA GEMM over nodes: C = A@Wp (+ A2@Wp2) + epilogue. Output N x 128.
// A staged in LDS (f16 fragment layout) once per block, shared by 8 waves.
// mode: 0 = +bias, 1 = +bias,BN,PReLU, 2 = +bias,PReLU.
// ---------------------------------------------------------------------------
__global__ __launch_bounds__(256) void gemm_wmma(
    const float* __restrict__ A,  int lda,  const _Float16* __restrict__ Wp,  int K,
    const float* __restrict__ A2, int lda2, const _Float16* __restrict__ Wp2, int K2,
    const float* __restrict__ bias,
    const float* __restrict__ bng, const float* __restrict__ bnb,
    const float* __restrict__ bnm, const float* __restrict__ bnv,
    const float* __restrict__ pa, int mode,
    float* __restrict__ C)
{
    __shared__ _Float16 sA[4096];               // 2 sources x (K/32<=4) x 512 halves

    int tm   = blockIdx.x;                      // row tile, 0..3124
    int tn   = threadIdx.y;                     // col tile, 0..7 (128 cols)
    int lane = threadIdx.x;
    int tid  = tn * 32 + lane;
    int row0 = tm * 16;

    // ---- cooperative stage of A (and A2) into f16 A-fragment layout --------
    {
        int nE = (K >> 5) * 512;
        for (int idx = tid; idx < nE; idx += 256) {
            int ks = idx >> 9, f = idx & 511, ln = f >> 4, j = f & 15;
            int r = ln & 15, kb = (ln >> 4) << 3;
            int k = (ks << 5) + ((j < 8) ? (kb + j) : (16 + kb + j - 8));
            sA[idx] = (_Float16)A[(size_t)(row0 + r) * lda + k];
        }
        if (A2) {
            int nE2 = (K2 >> 5) * 512;
            for (int idx = tid; idx < nE2; idx += 256) {
                int ks = idx >> 9, f = idx & 511, ln = f >> 4, j = f & 15;
                int r = ln & 15, kb = (ln >> 4) << 3;
                int k = (ks << 5) + ((j < 8) ? (kb + j) : (16 + kb + j - 8));
                sA[2048 + idx] = (_Float16)A2[(size_t)(row0 + r) * lda2 + k];
            }
        }
    }
    __syncthreads();

    // ---- WMMA main loops ---------------------------------------------------
    v8f acc = {0.f, 0.f, 0.f, 0.f, 0.f, 0.f, 0.f, 0.f};
    int nks = K >> 5;
    for (int ks = 0; ks < nks; ++ks) {
        v16h a = *(const v16h*)&sA[ks * 512 + lane * 16];
        v16h b = *(const v16h*)&Wp[(size_t)((ks * 8 + tn) * 32 + lane) * 16];
        acc = __builtin_amdgcn_wmma_f32_16x16x32_f16(false, a, false, b, (short)0, acc, false, false);
    }
    if (Wp2) {
        int nks2 = K2 >> 5;
        for (int ks = 0; ks < nks2; ++ks) {
            v16h a = *(const v16h*)&sA[2048 + ks * 512 + lane * 16];
            v16h b = *(const v16h*)&Wp2[(size_t)((ks * 8 + tn) * 32 + lane) * 16];
            acc = __builtin_amdgcn_wmma_f32_16x16x32_f16(false, a, false, b, (short)0, acc, false, false);
        }
    }

    // ---- epilogue ----------------------------------------------------------
    // D layout: element i -> row = row0 + i + (lane<16?0:8), col = tn*16 + (lane&15)
    int col   = tn * 16 + (lane & 15);
    int mbase = row0 + ((lane >> 4) << 3);
    float bb = bias ? bias[col] : 0.f;
    float g = 1.f, be = 0.f, mu = 0.f, iv = 1.f, al = 1.f;
    if (mode == 1) {
        g = bng[col]; be = bnb[col]; mu = bnm[col];
        iv = rsqrtf(bnv[col] + kEPS); al = pa[col];
    } else if (mode == 2) {
        al = pa[col];
    }
#pragma unroll
    for (int i = 0; i < 8; ++i) {
        float t = acc[i] + bb;
        if (mode == 1) { t = (t - mu) * iv * g + be; t = (t >= 0.f) ? t : al * t; }
        else if (mode == 2) { t = (t >= 0.f) ? t : al * t; }
        C[(size_t)(mbase + i) * HID + col] = t;
    }
}

// ---------------- helpers ------------------------------------------------------
__global__ void zero_f32(float* p, int n) {
    int i = blockIdx.x * 256 + threadIdx.x;
    if (i < n) p[i] = 0.f;
}

__global__ void deg_count(const int* __restrict__ ei, float* __restrict__ deg) {
    int e = blockIdx.x * 256 + threadIdx.x;
    if (e < EE) atomicAdd(&deg[ei[EE + e]], 1.f);
}

// one wave per edge, lanes stripe feature dim
__global__ void scatter_sum_edges(const float* __restrict__ X, const int* __restrict__ ei,
                                  float* __restrict__ acc, int dim) {
    int e = blockIdx.x * blockDim.y + threadIdx.y;
    if (e >= EE) return;
    int s = ei[e], d = ei[EE + e];
    const float* xs = X + (size_t)s * dim;
    float* ap = acc + (size_t)d * dim;
    for (int k = threadIdx.x; k < dim; k += 32) atomicAdd(&ap[k], xs[k]);
}

__global__ void divide_deg(float* __restrict__ acc, const float* __restrict__ deg, int dim) {
    int i = blockIdx.x * 256 + threadIdx.x;
    if (i < NN * dim) acc[i] /= fmaxf(deg[i / dim], 1.f);
}

// per-node attention scores: es = xw . a_src, ed = xw . a_dst (one wave per node)
__global__ void attn_scores(const float* __restrict__ xw, const float* __restrict__ as,
                            const float* __restrict__ ad, float* __restrict__ es,
                            float* __restrict__ ed) {
    int n = blockIdx.x * blockDim.y + threadIdx.y;
    if (n >= NN) return;
    const float* p = xw + (size_t)n * HID;
    float ss = 0.f, sd = 0.f;
    for (int c = threadIdx.x; c < HID; c += 32) { float v = p[c]; ss += v * as[c]; sd += v * ad[c]; }
#pragma unroll
    for (int o = 16; o > 0; o >>= 1) { ss += __shfl_xor(ss, o, 32); sd += __shfl_xor(sd, o, 32); }
    if (threadIdx.x == 0) { es[n] = ss; ed[n] = sd; }
}

// edge pass 1: w = exp(leaky(es[s]+ed[d])); den[d] += w   (softmax shift dropped: e is O(1))
__global__ void edge_weight(const int* __restrict__ ei, const float* __restrict__ es,
                            const float* __restrict__ ed, float* __restrict__ we,
                            float* __restrict__ den) {
    int e = blockIdx.x * 256 + threadIdx.x;
    if (e >= E2) return;
    int s, d;
    if (e < EE) { s = ei[e]; d = ei[EE + e]; } else { s = d = e - EE; }
    float t = es[s] + ed[d];
    t = (t >= 0.f) ? t : SLOPE * t;
    float w = __expf(t);
    we[e] = w;
    atomicAdd(&den[d], w);
}

// edge pass 2: acc[d] += (w/den[d]) * xw[s]   (one wave per edge)
__global__ void edge_scatter(const int* __restrict__ ei, const float* __restrict__ we,
                             const float* __restrict__ den, const float* __restrict__ xw,
                             float* __restrict__ acc) {
    int e = blockIdx.x * blockDim.y + threadIdx.y;
    if (e >= E2) return;
    int s, d;
    if (e < EE) { s = ei[e]; d = ei[EE + e]; } else { s = d = e - EE; }
    float coeff = we[e] / den[d];            // den > 0: every node has a self loop
    const float* xs = xw + (size_t)s * HID;
    float* ap = acc + (size_t)d * HID;
    for (int k = threadIdx.x; k < HID; k += 32) atomicAdd(&ap[k], coeff * xs[k]);
}

// mean over heads + bias + BN + PReLU, in place
__global__ void gat_finalize(float* __restrict__ acc, const float* __restrict__ bias,
                             const float* __restrict__ bng, const float* __restrict__ bnb,
                             const float* __restrict__ bnm, const float* __restrict__ bnv,
                             const float* __restrict__ pa) {
    int i = blockIdx.x * 256 + threadIdx.x;
    if (i >= NN * HID) return;
    int c = i & (HID - 1);
    float t = acc[i] * (1.f / HEADS) + bias[c];
    t = (t - bnm[c]) * rsqrtf(bnv[c] + kEPS) * bng[c] + bnb[c];
    acc[i] = (t >= 0.f) ? t : pa[c] * t;
}

// global mean pool: one wave per node
__global__ void pool_scatter(const float* __restrict__ h, const int* __restrict__ batch,
                             float* __restrict__ gsum, float* __restrict__ gcnt) {
    int n = blockIdx.x * blockDim.y + threadIdx.y;
    if (n >= NN) return;
    int b = batch[n];
    const float* p = h + (size_t)n * HID;
    float* g = gsum + (size_t)b * HID;
    for (int k = threadIdx.x; k < HID; k += 32) atomicAdd(&g[k], p[k]);
    if (threadIdx.x == 0) atomicAdd(&gcnt[b], 1.f);
}

__global__ void classifier(const float* __restrict__ gsum, const float* __restrict__ gcnt,
                           const float* __restrict__ Wc, const float* __restrict__ bc,
                           float* __restrict__ out) {
    int i = blockIdx.x * 256 + threadIdx.x;
    if (i >= GG * NC) return;
    int g = i / NC, c = i % NC;
    float inv = 1.f / fmaxf(gcnt[g], 1.f);
    const float* gp = gsum + (size_t)g * HID;
    float s = bc[c];
    for (int k = 0; k < HID; ++k) s += gp[k] * inv * Wc[k * NC + c];
    out[i] = s;
}

// ------------------------------------------------------------------------------
extern "C" void kernel_launch(void* const* d_in, const int* in_sizes, int n_in,
                              void* d_out, int out_size, void* d_ws, size_t ws_size,
                              hipStream_t stream) {
    const float* x      = (const float*)d_in[0];
    const int*   ei     = (const int*)  d_in[1];
    const int*   batch  = (const int*)  d_in[2];
    const float* s0_Wl  = (const float*)d_in[3];
    const float* s0_Wr  = (const float*)d_in[4];
    const float* s0_b   = (const float*)d_in[5];
    const float* s1_Wl  = (const float*)d_in[6];
    const float* s1_Wr  = (const float*)d_in[7];
    const float* s1_b   = (const float*)d_in[8];
    const float* bn_g   = (const float*)d_in[9];
    const float* bn_b   = (const float*)d_in[10];
    const float* bn_m   = (const float*)d_in[11];
    const float* bn_v   = (const float*)d_in[12];
    const float* pr_a   = (const float*)d_in[13];
    const float* g0_W   = (const float*)d_in[14];
    const float* g0_as  = (const float*)d_in[15];
    const float* g0_ad  = (const float*)d_in[16];
    const float* g0_bias= (const float*)d_in[17];
    const float* g1_W   = (const float*)d_in[18];
    const float* g1_as  = (const float*)d_in[19];
    const float* g1_ad  = (const float*)d_in[20];
    const float* g1_bias= (const float*)d_in[21];
    const float* idp_W  = (const float*)d_in[22];
    const float* idp_b  = (const float*)d_in[23];
    const float* idp_a  = (const float*)d_in[24];
    const float* cls_W  = (const float*)d_in[25];
    const float* cls_b  = (const float*)d_in[26];
    float* out = (float*)d_out;

    // workspace bump allocator
    char* wp = (char*)d_ws;
    auto alloc = [&](size_t bytes) -> char* {
        char* p = wp;
        wp += (bytes + 255) & ~(size_t)255;
        return p;
    };
    float* bufA = (float*)alloc((size_t)NN * HID * 4);
    float* bufB = (float*)alloc((size_t)NN * HID * 4);
    float* agg  = (float*)alloc((size_t)NN * HID * 4);  // also reused as xw per head
    float* deg  = (float*)alloc((size_t)NN * 4);
    float* es   = (float*)alloc((size_t)NN * 4);
    float* ed   = (float*)alloc((size_t)NN * 4);
    float* den  = (float*)alloc((size_t)NN * 4);
    float* we   = (float*)alloc((size_t)E2 * 4);
    float* gsum = (float*)alloc((size_t)GG * HID * 4);
    float* gcnt = (float*)alloc((size_t)GG * 4);
    // packed f16 weights (WMMA B-fragment layout)
    _Float16* p_s0Wl = (_Float16*)alloc((size_t)DIN * HID * 2);
    _Float16* p_s0Wr = (_Float16*)alloc((size_t)DIN * HID * 2);
    _Float16* p_s1Wl = (_Float16*)alloc((size_t)HID * HID * 2);
    _Float16* p_s1Wr = (_Float16*)alloc((size_t)HID * HID * 2);
    _Float16* p_g0   = (_Float16*)alloc((size_t)HEADS * HID * HID * 2);
    _Float16* p_g1   = (_Float16*)alloc((size_t)HEADS * HID * HID * 2);
    _Float16* p_idp  = (_Float16*)alloc((size_t)3 * HID * HID * 2);

    dim3 w8(32, 8);                         // 8 waves / block
    int gEdge8  = (EE + 7) / 8;
    int gE2_8   = (E2 + 7) / 8;
    int gNode8  = (NN + 7) / 8;
    int gNH     = (NN * HID + 255) / 256;
    int gN      = (NN + 255) / 256;
    int gPack64  = (DIN * HID + 255) / 256;
    int gPack128 = (HID * HID + 255) / 256;

    // ---- pack all weights to f16 fragment layout (tiny) --------------------
    pack_w_f16<<<gPack64, 256, 0, stream>>>(s0_Wl, HID, DIN, p_s0Wl);
    pack_w_f16<<<gPack64, 256, 0, stream>>>(s0_Wr, HID, DIN, p_s0Wr);
    pack_w_f16<<<gPack128, 256, 0, stream>>>(s1_Wl, HID, HID, p_s1Wl);
    pack_w_f16<<<gPack128, 256, 0, stream>>>(s1_Wr, HID, HID, p_s1Wr);
    for (int h = 0; h < HEADS; ++h) {
        pack_w_f16<<<gPack128, 256, 0, stream>>>(g0_W + h * HID, HEADS * HID, HID,
                                                 p_g0 + (size_t)h * HID * HID);
        pack_w_f16<<<gPack128, 256, 0, stream>>>(g1_W + h * HID, HEADS * HID, HID,
                                                 p_g1 + (size_t)h * HID * HID);
    }
    for (int l = 0; l < 3; ++l)
        pack_w_f16<<<gPack128, 256, 0, stream>>>(idp_W + (size_t)l * HID * HID, HID, HID,
                                                 p_idp + (size_t)l * HID * HID);

    // degree (shared by both SAGE layers)
    zero_f32<<<gN, 256, 0, stream>>>(deg, NN);
    deg_count<<<(EE + 255) / 256, 256, 0, stream>>>(ei, deg);

    // ---- SAGE0: h = mean_agg(x)@Wl + x@Wr + b -> BN0 -> PReLU0  (bufA)
    zero_f32<<<(NN * DIN + 255) / 256, 256, 0, stream>>>(agg, NN * DIN);
    scatter_sum_edges<<<gEdge8, w8, 0, stream>>>(x, ei, agg, DIN);
    divide_deg<<<(NN * DIN + 255) / 256, 256, 0, stream>>>(agg, deg, DIN);
    gemm_wmma<<<MTILES, w8, 0, stream>>>(agg, DIN, p_s0Wl, DIN,
                                         x, DIN, p_s0Wr, DIN,
                                         s0_b, bn_g, bn_b, bn_m, bn_v, pr_a, 1, bufA);

    // ---- SAGE1 (bufA -> bufB)
    zero_f32<<<gNH, 256, 0, stream>>>(agg, NN * HID);
    scatter_sum_edges<<<gEdge8, w8, 0, stream>>>(bufA, ei, agg, HID);
    divide_deg<<<gNH, 256, 0, stream>>>(agg, deg, HID);
    gemm_wmma<<<MTILES, w8, 0, stream>>>(agg, HID, p_s1Wl, HID,
                                         bufA, HID, p_s1Wr, HID,
                                         s1_b, bn_g + HID, bn_b + HID, bn_m + HID, bn_v + HID,
                                         pr_a + HID, 1, bufB);

    // ---- GAT0 (bufB -> bufA), head by head, xw in agg
    zero_f32<<<gNH, 256, 0, stream>>>(bufA, NN * HID);
    for (int h = 0; h < HEADS; ++h) {
        gemm_wmma<<<MTILES, w8, 0, stream>>>(bufB, HID, p_g0 + (size_t)h * HID * HID, HID,
                                             nullptr, 0, nullptr, 0,
                                             nullptr, nullptr, nullptr, nullptr, nullptr,
                                             nullptr, 0, agg);
        attn_scores<<<gNode8, w8, 0, stream>>>(agg, g0_as + h * HID, g0_ad + h * HID, es, ed);
        zero_f32<<<gN, 256, 0, stream>>>(den, NN);
        edge_weight<<<(E2 + 255) / 256, 256, 0, stream>>>(ei, es, ed, we, den);
        edge_scatter<<<gE2_8, w8, 0, stream>>>(ei, we, den, agg, bufA);
    }
    gat_finalize<<<gNH, 256, 0, stream>>>(bufA, g0_bias, bn_g + 2 * HID, bn_b + 2 * HID,
                                          bn_m + 2 * HID, bn_v + 2 * HID, pr_a + 2 * HID);

    // ---- GAT1 (bufA -> bufB)
    zero_f32<<<gNH, 256, 0, stream>>>(bufB, NN * HID);
    for (int h = 0; h < HEADS; ++h) {
        gemm_wmma<<<MTILES, w8, 0, stream>>>(bufA, HID, p_g1 + (size_t)h * HID * HID, HID,
                                             nullptr, 0, nullptr, 0,
                                             nullptr, nullptr, nullptr, nullptr, nullptr,
                                             nullptr, 0, agg);
        attn_scores<<<gNode8, w8, 0, stream>>>(agg, g1_as + h * HID, g1_ad + h * HID, es, ed);
        zero_f32<<<gN, 256, 0, stream>>>(den, NN);
        edge_weight<<<(E2 + 255) / 256, 256, 0, stream>>>(ei, es, ed, we, den);
        edge_scatter<<<gE2_8, w8, 0, stream>>>(ei, we, den, agg, bufB);
    }
    gat_finalize<<<gNH, 256, 0, stream>>>(bufB, g1_bias, bn_g + 3 * HID, bn_b + 3 * HID,
                                          bn_m + 3 * HID, bn_v + 3 * HID, pr_a + 3 * HID);

    // ---- IDP MLP: bufB -> bufA -> bufB -> bufA
    gemm_wmma<<<MTILES, w8, 0, stream>>>(bufB, HID, p_idp, HID,
                                         nullptr, 0, nullptr, 0,
                                         idp_b, nullptr, nullptr, nullptr, nullptr,
                                         idp_a, 2, bufA);
    gemm_wmma<<<MTILES, w8, 0, stream>>>(bufA, HID, p_idp + (size_t)HID * HID, HID,
                                         nullptr, 0, nullptr, 0,
                                         idp_b + HID, nullptr, nullptr, nullptr, nullptr,
                                         idp_a + HID, 2, bufB);
    gemm_wmma<<<MTILES, w8, 0, stream>>>(bufB, HID, p_idp + (size_t)2 * HID * HID, HID,
                                         nullptr, 0, nullptr, 0,
                                         idp_b + 2 * HID, nullptr, nullptr, nullptr, nullptr,
                                         nullptr, 0, bufA);

    // ---- global mean pool + classifier
    zero_f32<<<(GG * HID + 255) / 256, 256, 0, stream>>>(gsum, GG * HID);
    zero_f32<<<1, 256, 0, stream>>>(gcnt, GG);
    pool_scatter<<<gNode8, w8, 0, stream>>>(bufA, batch, gsum, gcnt);
    classifier<<<(GG * NC + 255) / 256, 256, 0, stream>>>(gsum, gcnt, cls_W, cls_b, out);

    (void)in_sizes; (void)n_in; (void)out_size; (void)ws_size;
}
